// GATWithMinCutPooling_83777632076182
// MI455X (gfx1250) — matrix operand order repaired and checked by reading, not compiled, not run on hardware
//
#include <hip/hip_runtime.h>
#include <math.h>
#include <stdint.h>

#define NN   8192       // nodes
#define NE   524288     // edges
#define NET  (NN + NE)  // edges incl. self loops = 532480
#define NIN  512
#define KC   64

typedef __attribute__((ext_vector_type(16))) __bf16 v16bf;
typedef __attribute__((ext_vector_type(8)))  float  v8f;

// ---------------- WMMA helpers (CDNA5 16x16x32 bf16, fp32 accum) ------------

__device__ __forceinline__ v8f wmma_bf16(v16bf a, v16bf b, v8f c) {
  return __builtin_amdgcn_wmma_f32_16x16x32_bf16(false, a, false, b,
                                                 (short)0, c, false, false);
}

// A-fragment k offset for element i, per ISA 16-bit A 16x32 layout
__device__ __forceinline__ int a_koff(int lane, int i) {
  return (i < 8) ? (((lane < 16) ? 0 : 8) + i)
                 : (((lane < 16) ? 16 : 24) + (i - 8));
}
// B-fragment k offset for element i, per ISA 16-bit B 32x16 layout
__device__ __forceinline__ int b_koff(int lane, int i) {
  return (i < 8) ? (((lane < 16) ? 0 : 8) + i)
                 : (((lane < 16) ? 16 : 24) + (i - 8));
}

__device__ __forceinline__ v16bf make_afrag(float4 p0, float4 p1,
                                            float4 p2, float4 p3) {
  v16bf a;
  a[0]  = (__bf16)p0.x; a[1]  = (__bf16)p0.y; a[2]  = (__bf16)p0.z; a[3]  = (__bf16)p0.w;
  a[4]  = (__bf16)p1.x; a[5]  = (__bf16)p1.y; a[6]  = (__bf16)p1.z; a[7]  = (__bf16)p1.w;
  a[8]  = (__bf16)p2.x; a[9]  = (__bf16)p2.y; a[10] = (__bf16)p2.z; a[11] = (__bf16)p2.w;
  a[12] = (__bf16)p3.x; a[13] = (__bf16)p3.y; a[14] = (__bf16)p3.z; a[15] = (__bf16)p3.w;
  return a;
}

// ---------------- CDNA5 async global->LDS staging ---------------------------
// Builtin signature (from compiler diagnostic): params are v4i* with AS quals.

typedef __attribute__((__vector_size__(16))) int v4i;
typedef __attribute__((address_space(1))) v4i gv4i;
typedef __attribute__((address_space(3))) v4i lv4i;

__device__ __forceinline__ void stage16(const float* g, float* l) {
#if __has_builtin(__builtin_amdgcn_global_load_async_to_lds_b128)
  __builtin_amdgcn_global_load_async_to_lds_b128((gv4i*)g, (lv4i*)l, 0, 0);
#else
  *(float4*)l = *(const float4*)g;
#endif
}

__device__ __forceinline__ void stage_wait() {
#if __has_builtin(__builtin_amdgcn_global_load_async_to_lds_b128)
#if __has_builtin(__builtin_amdgcn_s_wait_asynccnt)
  __builtin_amdgcn_s_wait_asynccnt(0);
#else
  asm volatile("s_wait_asynccnt 0x0" ::: "memory");
#endif
#endif
}

// ---------------- weight packing: B[K,64] -> per-fragment bf16 --------------
// Bp layout: [kt][tn][lane][i], 16 bf16 (32B) contiguous per lane fragment.

__global__ void pack_b_kernel(const float* __restrict__ B,
                              __bf16* __restrict__ Bp, int ktiles) {
  int t = blockIdx.x * blockDim.x + threadIdx.x;   // one fragment per thread
  if (t >= ktiles * 4 * 32) return;
  int lane = t & 31, tn = (t >> 5) & 3, kt = t >> 7;
  const float* src = B + (size_t)(kt * 32) * 64 + tn * 16 + (lane & 15);
  __bf16* dst = Bp + (size_t)t * 16;
#pragma unroll
  for (int i = 0; i < 16; ++i) dst[i] = (__bf16)src[b_koff(lane, i) * 64];
}

// ---------------- GEMM: C[NN,64] = A[NN,K] x B[K,64], packed-B WMMA ---------

template <int K>
__global__ void gemm_bf16_packed(const float* __restrict__ A,
                                 const __bf16* __restrict__ Bp,
                                 float* __restrict__ C) {
  int wid  = (blockIdx.x * blockDim.x + threadIdx.x) >> 5;
  int lane = threadIdx.x & 31;
  int tm = wid >> 2, tn = wid & 3;            // 4 n-tiles (N=64)
  int arow = tm * 16 + (lane & 15);
  const float4* Ar = (const float4*)(A + (size_t)arow * K);
  int ao = (lane < 16) ? 0 : 2;
  v8f acc = {};
#pragma unroll
  for (int kt = 0; kt < K / 32; ++kt) {
    float4 p0 = Ar[kt * 8 + ao + 0];
    float4 p1 = Ar[kt * 8 + ao + 1];
    float4 p2 = Ar[kt * 8 + ao + 4];
    float4 p3 = Ar[kt * 8 + ao + 5];
    v16bf a = make_afrag(p0, p1, p2, p3);
    v16bf b = *(const v16bf*)(Bp + (size_t)(((kt * 4 + tn) * 32 + lane)) * 16);
    acc = wmma_bf16(a, b, acc);
  }
  float* Cr = C + (size_t)(tm * 16 + ((lane < 16) ? 0 : 8)) * 64
                + tn * 16 + (lane & 15);
#pragma unroll
  for (int v = 0; v < 8; ++v) Cr[v * 64] = acc[v];
}

// ---------------- small utility kernels ------------------------------------

__global__ void fill_kernel(float* __restrict__ p, float v, int n) {
  int i = blockIdx.x * blockDim.x + threadIdx.x;
  if (i < n) p[i] = v;
}

__device__ __forceinline__ float lrelu(float v) { return v > 0.f ? v : 0.2f * v; }

__device__ __forceinline__ void atomicMaxF(float* addr, float val) {
  if (val >= 0.f) atomicMax((int*)addr, __float_as_int(val));
  else            atomicMin((unsigned int*)addr, __float_as_uint(val));
}

// ---------------- GAT layer 1 (heads=8, ch=8) -------------------------------

__global__ void att_prep8_kernel(const float* __restrict__ h,
                                 const float* __restrict__ asrc,
                                 const float* __restrict__ adst,
                                 float* __restrict__ als,
                                 float* __restrict__ ald) {
  int t = blockIdx.x * blockDim.x + threadIdx.x;   // t = n*8 + head
  if (t >= NN * 8) return;
  int n = t >> 3, hh = t & 7;
  const float* row = h + (size_t)n * 64 + hh * 8;
  float s = 0.f, d = 0.f;
#pragma unroll
  for (int c = 0; c < 8; ++c) {
    s += row[c] * asrc[hh * 8 + c];
    d += row[c] * adst[hh * 8 + c];
  }
  als[t] = s; ald[t] = d;
}

__global__ void edge_max8_kernel(const int* __restrict__ ei,
                                 const float* __restrict__ als,
                                 const float* __restrict__ ald,
                                 float* __restrict__ emax) {
  int e = blockIdx.x * blockDim.x + threadIdx.x;
  if (e >= NET) return;
  int s, d;
  if (e < NE) { s = ei[e]; d = ei[NE + e]; } else { s = d = e - NE; }
#pragma unroll
  for (int hh = 0; hh < 8; ++hh)
    atomicMaxF(&emax[d * 8 + hh], lrelu(als[s * 8 + hh] + ald[d * 8 + hh]));
}

__global__ void edge_sum8_kernel(const int* __restrict__ ei,
                                 const float* __restrict__ als,
                                 const float* __restrict__ ald,
                                 const float* __restrict__ emax,
                                 float* __restrict__ esum) {
  int e = blockIdx.x * blockDim.x + threadIdx.x;
  if (e >= NET) return;
  int s, d;
  if (e < NE) { s = ei[e]; d = ei[NE + e]; } else { s = d = e - NE; }
#pragma unroll
  for (int hh = 0; hh < 8; ++hh) {
    float v = lrelu(als[s * 8 + hh] + ald[d * 8 + hh]);
    atomicAdd(&esum[d * 8 + hh], __expf(v - emax[d * 8 + hh]));
  }
}

__global__ void edge_scatter8_kernel(const int* __restrict__ ei,
                                     const float* __restrict__ h,
                                     const float* __restrict__ als,
                                     const float* __restrict__ ald,
                                     const float* __restrict__ emax,
                                     const float* __restrict__ esum,
                                     float* __restrict__ agg) {
  int t = blockIdx.x * blockDim.x + threadIdx.x;   // t = e*8 + head
  if (t >= NET * 8) return;
  int e = t >> 3, hh = t & 7;
  int s, d;
  if (e < NE) { s = ei[e]; d = ei[NE + e]; } else { s = d = e - NE; }
  float v = lrelu(als[s * 8 + hh] + ald[d * 8 + hh]);
  float alpha = __expf(v - emax[d * 8 + hh]) / (esum[d * 8 + hh] + 1e-16f);
  const float* hs = h + (size_t)s * 64 + hh * 8;
  float* ag = agg + (size_t)d * 64 + hh * 8;
#pragma unroll
  for (int c = 0; c < 8; ++c) atomicAdd(&ag[c], hs[c] * alpha);
}

__global__ void bias_elu_kernel(float* __restrict__ agg,
                                const float* __restrict__ b) {
  int i = blockIdx.x * blockDim.x + threadIdx.x;
  if (i >= NN * 64) return;
  float v = agg[i] + b[i & 63];
  agg[i] = v > 0.f ? v : expm1f(v);
}

// ---------------- GAT layer 2 (heads=1, ch=64) ------------------------------

__global__ void att_prep1_kernel(const float* __restrict__ h,
                                 const float* __restrict__ asrc,
                                 const float* __restrict__ adst,
                                 float* __restrict__ als,
                                 float* __restrict__ ald) {
  int n = blockIdx.x * blockDim.x + threadIdx.x;
  if (n >= NN) return;
  const float* row = h + (size_t)n * 64;
  float s = 0.f, d = 0.f;
#pragma unroll 8
  for (int c = 0; c < 64; ++c) { s += row[c] * asrc[c]; d += row[c] * adst[c]; }
  als[n] = s; ald[n] = d;
}

__global__ void edge_max1_kernel(const int* __restrict__ ei,
                                 const float* __restrict__ als,
                                 const float* __restrict__ ald,
                                 float* __restrict__ emax) {
  int e = blockIdx.x * blockDim.x + threadIdx.x;
  if (e >= NET) return;
  int s, d;
  if (e < NE) { s = ei[e]; d = ei[NE + e]; } else { s = d = e - NE; }
  atomicMaxF(&emax[d], lrelu(als[s] + ald[d]));
}

__global__ void edge_sum1_kernel(const int* __restrict__ ei,
                                 const float* __restrict__ als,
                                 const float* __restrict__ ald,
                                 const float* __restrict__ emax,
                                 float* __restrict__ esum) {
  int e = blockIdx.x * blockDim.x + threadIdx.x;
  if (e >= NET) return;
  int s, d;
  if (e < NE) { s = ei[e]; d = ei[NE + e]; } else { s = d = e - NE; }
  atomicAdd(&esum[d], __expf(lrelu(als[s] + ald[d]) - emax[d]));
}

__global__ void edge_scatter1_kernel(const int* __restrict__ ei,
                                     const float* __restrict__ h,
                                     const float* __restrict__ als,
                                     const float* __restrict__ ald,
                                     const float* __restrict__ emax,
                                     const float* __restrict__ esum,
                                     float* __restrict__ agg) {
  int t = blockIdx.x * blockDim.x + threadIdx.x;   // t = e*8 + chunk(8ch)
  if (t >= NET * 8) return;
  int e = t >> 3, q = (t & 7) * 8;
  int s, d;
  if (e < NE) { s = ei[e]; d = ei[NE + e]; } else { s = d = e - NE; }
  float alpha = __expf(lrelu(als[s] + ald[d]) - emax[d]) / (esum[d] + 1e-16f);
  const float* hs = h + (size_t)s * 64 + q;
  float* ag = agg + (size_t)d * 64 + q;
#pragma unroll
  for (int c = 0; c < 8; ++c) atomicAdd(&ag[c], hs[c] * alpha);
}

// ---------------- double row-softmax:  s = sm(x2),  s2 = sm(s) --------------

__global__ void softmax2_kernel(float* __restrict__ x2,
                                const float* __restrict__ b2,
                                float* __restrict__ s_out,
                                float* __restrict__ s2_out) {
  int n = blockIdx.x * blockDim.x + threadIdx.x;
  if (n >= NN) return;
  float* row = x2 + (size_t)n * 64;
  float v[64];
  float m = -3.4e38f;
#pragma unroll
  for (int c = 0; c < 64; ++c) {
    float t = row[c] + b2[c];
    row[c] = t;                       // keep x2 (with bias) for x_pool
    v[c] = t; m = fmaxf(m, t);
  }
  float sum = 0.f;
#pragma unroll
  for (int c = 0; c < 64; ++c) { v[c] = __expf(v[c] - m); sum += v[c]; }
  float inv = 1.f / sum, m2 = -3.4e38f;
#pragma unroll
  for (int c = 0; c < 64; ++c) {
    v[c] *= inv;
    s_out[(size_t)n * 64 + c] = v[c];
    m2 = fmaxf(m2, v[c]);
  }
  float sum2 = 0.f;
#pragma unroll
  for (int c = 0; c < 64; ++c) { v[c] = __expf(v[c] - m2); sum2 += v[c]; }
  float inv2 = 1.f / sum2;
#pragma unroll
  for (int c = 0; c < 64; ++c) s2_out[(size_t)n * 64 + c] = v[c] * inv2;
}

// ---------------- x_pool = s2^T @ x2  (64x8192 . 8192x64), WMMA, 1 block ----

__global__ void pool_gemm_kernel(const float* __restrict__ s2,
                                 const float* __restrict__ x2,
                                 float* __restrict__ out) {
  __shared__ float lsa[32 * 64];   // s2 rows k0..k0+31
  __shared__ float lsb[32 * 64];   // x2 rows k0..k0+31
  int t = threadIdx.x;             // 512 threads = 16 waves
  int wid = t >> 5, lane = t & 31;
  int tm = wid >> 2, tn = wid & 3;
  int mcol = tm * 16 + (lane & 15);
  int ncol = tn * 16 + (lane & 15);
  v8f acc = {};

  for (int k0 = 0; k0 < NN; k0 += 32) {
    stage16(s2 + (size_t)k0 * 64 + t * 4, lsa + t * 4);
    stage16(x2 + (size_t)k0 * 64 + t * 4, lsb + t * 4);
    stage_wait();
    __syncthreads();
    v16bf a, b;
#pragma unroll
    for (int i = 0; i < 16; ++i) a[i] = (__bf16)lsa[a_koff(lane, i) * 64 + mcol];
#pragma unroll
    for (int i = 0; i < 16; ++i) b[i] = (__bf16)lsb[b_koff(lane, i) * 64 + ncol];
    acc = wmma_bf16(a, b, acc);
    __syncthreads();
  }
#pragma unroll
  for (int v = 0; v < 8; ++v) {
    int m = tm * 16 + v + ((lane < 16) ? 0 : 8);
    out[m * 64 + ncol] = acc[v];
  }
}

// -------- out_adj[a,b] = sum_e s2[src_e,a]*s2[dst_e,b]  (gathered WMMA) -----
// 256 blocks x 512 threads; each block reduces 2048 raw edges, atomics at end.

__global__ void outadj_kernel(const int* __restrict__ ei,
                              const float* __restrict__ s2,
                              float* __restrict__ outadj) {
  __shared__ int   sid[32], did[32];
  __shared__ float lsa[32 * 64];   // s2[src] rows, [edge][cluster]
  __shared__ float lsb[32 * 64];   // s2[dst] rows
  int t = threadIdx.x;
  int wid = t >> 5, lane = t & 31;
  int tm = wid >> 2, tn = wid & 3;
  int mcol = tm * 16 + (lane & 15);
  int ncol = tn * 16 + (lane & 15);
  int base = blockIdx.x * 2048;
  v8f acc = {};

  for (int g = 0; g < 64; ++g) {
    __syncthreads();                         // previous frag reads done
    if (t < 32) {
      int e = base + g * 32 + t;
      sid[t] = ei[e];
      did[t] = ei[NE + e];
    }
    __syncthreads();
    {
      int r = t >> 4, c4 = (t & 15) * 4;     // 512 float4 slots per array
      stage16(s2 + (size_t)sid[r] * 64 + c4, lsa + r * 64 + c4);
      stage16(s2 + (size_t)did[r] * 64 + c4, lsb + r * 64 + c4);
    }
    stage_wait();
    __syncthreads();
    v16bf a, b;
#pragma unroll
    for (int i = 0; i < 16; ++i) a[i] = (__bf16)lsa[a_koff(lane, i) * 64 + mcol];
#pragma unroll
    for (int i = 0; i < 16; ++i) b[i] = (__bf16)lsb[b_koff(lane, i) * 64 + ncol];
    acc = wmma_bf16(a, b, acc);
  }
#pragma unroll
  for (int v = 0; v < 8; ++v) {
    int m = tm * 16 + v + ((lane < 16) ? 0 : 8);
    atomicAdd(&outadj[m * 64 + ncol], acc[v]);
  }
}

// ---------------- finalize: zero diag, degree normalize ---------------------

__global__ void finalize_kernel(const float* __restrict__ oa,
                                float* __restrict__ adjp) {
  __shared__ float d[64];
  int a = threadIdx.x;               // 64 threads
  float sum = 0.f;
  for (int b = 0; b < 64; ++b) sum += (b == a) ? 0.f : oa[a * 64 + b];
  d[a] = sqrtf(sum) + 1e-15f;
  __syncthreads();
  for (int b = 0; b < 64; ++b) {
    float v = (b == a) ? 0.f : oa[a * 64 + b];
    adjp[a * 64 + b] = v / (d[a] * d[b]);
  }
}

// ---------------------------------------------------------------------------

extern "C" void kernel_launch(void* const* d_in, const int* in_sizes, int n_in,
                              void* d_out, int out_size, void* d_ws, size_t ws_size,
                              hipStream_t stream) {
  const float* x   = (const float*)d_in[0];
  const int*   ei  = (const int*)  d_in[1];
  const float* W1  = (const float*)d_in[2];
  const float* as1 = (const float*)d_in[3];
  const float* ad1 = (const float*)d_in[4];
  const float* b1  = (const float*)d_in[5];
  const float* W2  = (const float*)d_in[6];
  const float* as2 = (const float*)d_in[7];
  const float* ad2 = (const float*)d_in[8];
  const float* b2  = (const float*)d_in[9];
  float* out = (float*)d_out;
  float* ws  = (float*)d_ws;

  // workspace layout (floats)
  float* h1     = ws;                    // [NN*64]  h1; later reused as agg2/x2
  float* agg1   = ws + 524288;           // [NN*64]  agg1/hout; later s2
  float* h2     = ws + 1048576;          // [NN*64]  layer-2 features
  float* als1   = ws + 1572864;          // [NN*8]
  float* ald1   = ws + 1638400;          // [NN*8]
  float* emax1  = ws + 1703936;          // [NN*8]
  float* esum1  = ws + 1769472;          // [NN*8]
  float* als2   = ws + 1835008;          // [NN]
  float* ald2   = ws + 1843200;          // [NN]
  float* emax2  = ws + 1851392;          // [NN]
  float* esum2  = ws + 1859584;          // [NN]
  float* outadj = ws + 1867776;          // [64*64]
  __bf16* Bp1   = (__bf16*)(ws + 1871872); // packed W1: 32768 bf16 (64KB)
  __bf16* Bp2   = (__bf16*)(ws + 1888256); // packed W2: 4096 bf16 (8KB)
  float* x2 = h1;                        // alias after layer 1 complete
  float* s2 = agg1;                      // alias after GEMM2 complete

  const float NEGINF = -3.402823466e38f;

  // ---- pack weights into bf16 B-fragment layout ----
  pack_b_kernel<<<8, 256, 0, stream>>>(W1, Bp1, NIN / 32);
  pack_b_kernel<<<1, 256, 0, stream>>>(W2, Bp2, 64 / 32);

  // ---- layer 1: h1 = x @ W1  (8192x512x64, WMMA) ----
  gemm_bf16_packed<NIN><<<256, 256, 0, stream>>>(x, Bp1, h1);

  fill_kernel<<<256,  256, 0, stream>>>(emax1, NEGINF, NN * 8);
  fill_kernel<<<256,  256, 0, stream>>>(esum1, 0.f,    NN * 8);
  fill_kernel<<<2048, 256, 0, stream>>>(agg1,  0.f,    NN * 64);
  fill_kernel<<<32,   256, 0, stream>>>(emax2, NEGINF, NN);
  fill_kernel<<<32,   256, 0, stream>>>(esum2, 0.f,    NN);
  fill_kernel<<<16,   256, 0, stream>>>(outadj, 0.f,   64 * 64);

  att_prep8_kernel<<<256, 256, 0, stream>>>(h1, as1, ad1, als1, ald1);
  edge_max8_kernel<<<NET / 256, 256, 0, stream>>>(ei, als1, ald1, emax1);
  edge_sum8_kernel<<<NET / 256, 256, 0, stream>>>(ei, als1, ald1, emax1, esum1);
  edge_scatter8_kernel<<<NET * 8 / 256, 256, 0, stream>>>(ei, h1, als1, ald1,
                                                          emax1, esum1, agg1);
  bias_elu_kernel<<<2048, 256, 0, stream>>>(agg1, b1);   // agg1 -> hout

  // ---- layer 2: h2 = hout @ W2  (8192x64x64, WMMA) ----
  gemm_bf16_packed<64><<<256, 256, 0, stream>>>(agg1, Bp2, h2);

  att_prep1_kernel<<<32, 256, 0, stream>>>(h2, as2, ad2, als2, ald2);
  fill_kernel<<<2048, 256, 0, stream>>>(x2, 0.f, NN * 64);   // agg2 (aliases h1)
  edge_max1_kernel<<<NET / 256, 256, 0, stream>>>(ei, als2, ald2, emax2);
  edge_sum1_kernel<<<NET / 256, 256, 0, stream>>>(ei, als2, ald2, emax2, esum2);
  edge_scatter1_kernel<<<NET * 8 / 256, 256, 0, stream>>>(ei, h2, als2, ald2,
                                                          emax2, esum2, x2);

  // ---- softmax twice: s -> d_out[8192..], s2 -> ws ----
  softmax2_kernel<<<32, 256, 0, stream>>>(x2, b2, out + 8192, s2);

  // ---- x_pool = s2^T @ x2 -> d_out[0..4095]  (WMMA) ----
  pool_gemm_kernel<<<1, 512, 0, stream>>>(s2, x2, out);

  // ---- out_adj via edge-gathered WMMA, then normalize -> d_out[4096..8191] ----
  outadj_kernel<<<256, 512, 0, stream>>>(ei, s2, outadj);
  finalize_kernel<<<1, 64, 0, stream>>>(outadj, out + 4096);
}
